// Attn__73478300500251
// MI455X (gfx1250) — matrix operand order repaired and checked by reading, not compiled, not run on hardware
//
#include <hip/hip_runtime.h>
#include <hip/hip_bf16.h>

// ---------------------------------------------------------------------------
// MaxViT grid-attention block for MI455X (gfx1250), wave32 + WMMA f16 +
// async-to-LDS double-buffered GEMM pipeline.
// B=16 H=64 W=80 C=256, windows: 1024 x 80 tokens, heads=8, dh=32, inner=1024.
// ---------------------------------------------------------------------------

typedef _Float16 v16h __attribute__((ext_vector_type(16)));
typedef _Float16 h8   __attribute__((ext_vector_type(8)));
typedef float    v8f  __attribute__((ext_vector_type(8)));

union V16U { v16h v; h8 h[2]; };

__device__ __forceinline__ v8f wmma16(v16h a, v16h b, v8f c) {
  // D = A(16x32 f16) * B(32x16 f16) + C(16x16 f32)
  return __builtin_amdgcn_wmma_f32_16x16x32_f16(false, a, false, b, (short)0, c,
                                                false, false);
}

// Truncate a generic pointer derived from __shared__ to its LDS byte offset
// (flat LDS aperture keeps the workgroup-relative offset in the low 32 bits).
__device__ __forceinline__ unsigned lds_off(const void* p) {
  return (unsigned)(size_t)p;
}

// CDNA5 async DMA: copy 16B from global to LDS, tracked by ASYNCcnt.
__device__ __forceinline__ void async_b128(unsigned lds, const void* g) {
  asm volatile("global_load_async_to_lds_b128 %0, %1, off"
               :: "v"(lds), "v"((unsigned long long)(size_t)g)
               : "memory");
}
__device__ __forceinline__ void wait_async0() {
  asm volatile("s_wait_asynccnt 0x0" ::: "memory");
}

// A fragment: lane(0-15) holds row M=lane, K chunks [0..7],[16..23];
//             lane(16-31) holds row M=lane-16, K chunks [8..15],[24..31].
// ldk = row pitch in elements.
__device__ __forceinline__ v16h load_a_frag(const _Float16* A, int ldk, int l16, int lh) {
  const _Float16* p = A + l16 * ldk + lh * 8;
  V16U u;
  u.h[0] = *(const h8*)(p);
  u.h[1] = *(const h8*)(p + 16);
  return u.v;
}

// B fragment from B^T (row n = lane%16, K chunks [8h..8h+7],[16+8h..16+8h+7], h=lane/16).
__device__ __forceinline__ v16h load_b_frag(const _Float16* Bt, int ldk, int l16, int lh) {
  const _Float16* p = Bt + l16 * ldk + lh * 8;
  V16U u;
  u.h[0] = *(const h8*)(p);
  u.h[1] = *(const h8*)(p + 16);
  return u.v;
}

// ---------------------------------------------------------------------------
// Block GEMM core: 256 threads (8 waves), block tile 128(M) x 128(N),
// wave tile 64x32 (4x2 WMMA tiles). K-slices of A and B^T (128x32 f16) are
// async-staged into double-buffered LDS (row pitch 56 halves = 112B).
// ---------------------------------------------------------------------------
#define LDS_PITCH 56

__device__ __forceinline__ void stage_panel(const _Float16* __restrict__ g,
                                            unsigned lbase, int K, int k0, int tid) {
  // 128 rows x 32 halves = 512 x 16B chunks; 256 threads -> 2 chunks each
  for (int t = 0; t < 2; ++t) {
    int c = tid + t * 256;
    int row = c >> 2, kc = (c & 3) * 8;
    const _Float16* src = g + (size_t)row * K + k0 + kc;
    async_b128(lbase + (unsigned)(row * LDS_PITCH + kc) * 2, src);
    __builtin_prefetch(src + 64, 0, 0);   // prime L2 two K-steps ahead
  }
}

__device__ __forceinline__ void gemm_core(const _Float16* __restrict__ Ag,
                                          const _Float16* __restrict__ Bg,
                                          int K, v8f c[4][2]) {
  __shared__ __align__(16) _Float16 sA[2][128 * LDS_PITCH];
  __shared__ __align__(16) _Float16 sB[2][128 * LDS_PITCH];
  int tid = threadIdx.x;
  int lane = tid & 31, l16 = lane & 15, lh = lane >> 4;
  int wid = tid >> 5, wm = wid & 1, wn = wid >> 1;
  int m0w = wm * 64, n0w = wn * 32;
  unsigned la[2] = {lds_off(sA[0]), lds_off(sA[1])};
  unsigned lb[2] = {lds_off(sB[0]), lds_off(sB[1])};
  int KS = K / 32;

  stage_panel(Ag, la[0], K, 0, tid);
  stage_panel(Bg, lb[0], K, 0, tid);
  wait_async0();
  __syncthreads();

  int buf = 0;
  for (int ks = 0; ks < KS; ++ks) {
    if (ks + 1 < KS) {                       // overlap next-tile DMA with math
      stage_panel(Ag, la[buf ^ 1], K, (ks + 1) * 32, tid);
      stage_panel(Bg, lb[buf ^ 1], K, (ks + 1) * 32, tid);
    }
    const _Float16* Ash = sA[buf];
    const _Float16* Bsh = sB[buf];
    v16h b0 = load_b_frag(Bsh + n0w * LDS_PITCH, LDS_PITCH, l16, lh);
    v16h b1 = load_b_frag(Bsh + (n0w + 16) * LDS_PITCH, LDS_PITCH, l16, lh);
    for (int mf = 0; mf < 4; ++mf) {
      v16h af = load_a_frag(Ash + (m0w + mf * 16) * LDS_PITCH, LDS_PITCH, l16, lh);
      c[mf][0] = wmma16(af, b0, c[mf][0]);
      c[mf][1] = wmma16(af, b1, c[mf][1]);
    }
    wait_async0();                           // next-tile DMA done
    __syncthreads();                         // everyone done reading this tile
    buf ^= 1;
  }
}

// --------------------------- weight prep -----------------------------------
// fp32 [K][N] row-major -> f16 [N][K] (B^T layout for WMMA B fragments)
__global__ void k_transpose_cast(const float* __restrict__ src,
                                 _Float16* __restrict__ dst, int K, int N) {
  int i = blockIdx.x * 256 + threadIdx.x;
  if (i >= K * N) return;
  int n = i / K, k = i - n * K;
  dst[i] = (_Float16)src[k * N + n];
}

// --------------------------- LayerNorm -------------------------------------
// One wave per token (32 lanes x 8 channels). permute!=0 applies the
// grid-partition reorder: pixel (b,h,w) -> token t = window*80 + n with
// window=(b*8+h%8)*8+w%8, n=(h/8)*10+(w/8).
__global__ void __launch_bounds__(256) k_layernorm(const float* __restrict__ x,
                                                   const float* __restrict__ g,
                                                   const float* __restrict__ bta,
                                                   _Float16* __restrict__ out,
                                                   int permute) {
  int wid = threadIdx.x >> 5, lane = threadIdx.x & 31;
  int p = blockIdx.x * 8 + wid;
  const float* row = x + (size_t)p * 256;
  float4 v0 = *(const float4*)(row + lane * 8);
  float4 v1 = *(const float4*)(row + lane * 8 + 4);
  float vals[8] = {v0.x, v0.y, v0.z, v0.w, v1.x, v1.y, v1.z, v1.w};
  float s = 0.f, sq = 0.f;
  for (int i = 0; i < 8; ++i) { s += vals[i]; sq += vals[i] * vals[i]; }
  for (int off = 16; off > 0; off >>= 1) {
    s += __shfl_xor(s, off, 32);
    sq += __shfl_xor(sq, off, 32);
  }
  float mean = s * (1.f / 256.f);
  float var  = sq * (1.f / 256.f) - mean * mean;
  float rs   = rsqrtf(var + 1e-5f);
  int t = p;
  if (permute) {
    int bb = p / 5120, r = p - bb * 5120;
    int hh = r / 80, ww = r - hh * 80;
    int gh = hh >> 3, h2 = hh & 7, gw = ww >> 3, w2 = ww & 7;
    t = ((bb * 8 + h2) * 8 + w2) * 80 + gh * 10 + gw;
  }
  h8 o;
  for (int i = 0; i < 8; ++i) {
    int cidx = lane * 8 + i;
    o[i] = (_Float16)((vals[i] - mean) * rs * g[cidx] + bta[cidx]);
  }
  *(h8*)(out + (size_t)t * 256 + lane * 8) = o;
}

// --------------------------- qkv GEMM --------------------------------------
// [T,256]f16 @ [256,768] + b -> scatter into q[nW,8,80,32], k[nW,8,80,32],
// v^T[nW,8,32,96] (cols 80..95 never read because P is zero-padded).
__global__ void __launch_bounds__(256) k_gemm_qkv(const _Float16* __restrict__ A,
                                                  const _Float16* __restrict__ Bt,
                                                  const float* __restrict__ bias,
                                                  _Float16* __restrict__ q,
                                                  _Float16* __restrict__ kb,
                                                  _Float16* __restrict__ vt) {
  int mblk = blockIdx.x * 128, nblk = blockIdx.y * 128;
  v8f c[4][2] = {};
  gemm_core(A + (size_t)mblk * 256, Bt + (size_t)nblk * 256, 256, c);
  int lane = threadIdx.x & 31, l16 = lane & 15, lh = lane >> 4;
  int wid = threadIdx.x >> 5, wm = wid & 1, wn = wid >> 1;
  for (int mf = 0; mf < 4; ++mf)
    for (int nf = 0; nf < 2; ++nf) {
      int n = nblk + wn * 32 + nf * 16 + l16;
      float bv = bias[n];
      int head = n / 96, jj = n - head * 96;
      for (int j = 0; j < 8; ++j) {
        int m = mblk + wm * 64 + mf * 16 + j + lh * 8;
        float val = c[mf][nf][j] + bv;
        int w = m / 80, nn = m - w * 80;
        int base = ((w * 8 + head) * 80 + nn) * 32;
        if (jj < 32)      q[base + jj] = (_Float16)val;
        else if (jj < 64) kb[base + jj - 32] = (_Float16)val;
        else              vt[((w * 8 + head) * 32 + (jj - 64)) * 96 + nn] = (_Float16)val;
      }
    }
}

// --------------------------- attention -------------------------------------
// One block per (window, head); 5 waves, wave r owns S/O rows [16r,16r+16).
// K and V^T tiles are async-staged into LDS (shared by all 5 waves).
__global__ void __launch_bounds__(160) k_attention(const _Float16* __restrict__ q,
                                                   const _Float16* __restrict__ kb,
                                                   const _Float16* __restrict__ vt,
                                                   _Float16* __restrict__ o) {
  __shared__ __align__(16) float    Ss[80 * 81];          // stride 81: bank spread
  __shared__ __align__(16) _Float16 Ps[80 * 96];          // f16 probs, K pad to 96
  __shared__ __align__(16) _Float16 Ks[80 * LDS_PITCH];   // K tile 80x32
  __shared__ __align__(16) _Float16 Vs[32 * 96];          // V^T tile 32x96
  int w = blockIdx.x >> 3, head = blockIdx.x & 7;
  const _Float16* Q  = q  + (size_t)((w * 8 + head) * 80) * 32;
  const _Float16* Km = kb + (size_t)((w * 8 + head) * 80) * 32;
  const _Float16* Vt = vt + (size_t)((w * 8 + head) * 32) * 96;
  int tid = threadIdx.x;
  int wave = tid >> 5, lane = tid & 31;
  int l16 = lane & 15, lh = lane >> 4;

  // Async stage: K (80 rows x 64B = 320 chunks), V^T (32 rows x 192B = 384 chunks)
  unsigned kbase = lds_off(Ks), vbase = lds_off(Vs);
  for (int t = 0; t < 2; ++t) {
    int cc = tid + t * 160;
    int row = cc >> 2, kc = (cc & 3) * 8;
    async_b128(kbase + (unsigned)(row * LDS_PITCH + kc) * 2, Km + row * 32 + kc);
  }
  for (int t = 0; t < 3; ++t) {
    int cc = tid + t * 160;
    if (cc < 384) {
      int row = cc / 12, kc = (cc % 12) * 8;
      async_b128(vbase + (unsigned)(row * 96 + kc) * 2, Vt + row * 96 + kc);
    }
  }
  wait_async0();
  __syncthreads();

  // S = (Q K^T) * scale  (K-dim = 32: one WMMA step; 5 column tiles)
  v16h a = load_a_frag(Q + wave * 16 * 32, 32, l16, lh);
  for (int ct = 0; ct < 5; ++ct) {
    v16h bfr = load_b_frag(Ks + ct * 16 * LDS_PITCH, LDS_PITCH, l16, lh);
    v8f c = {};
    c = wmma16(a, bfr, c);
    for (int j = 0; j < 8; ++j) {
      int m = wave * 16 + j + lh * 8;
      Ss[m * 81 + ct * 16 + l16] = c[j] * 0.17677669529663687f; // 32^-0.5
    }
  }
  __syncthreads();

  // softmax: one thread per row (threads 0..79)
  int r = tid;
  if (r < 80) {
    float mx = -3.4e38f;
    for (int i = 0; i < 80; ++i) mx = fmaxf(mx, Ss[r * 81 + i]);
    float sum = 0.f;
    for (int i = 0; i < 80; ++i) {
      float e = __expf(Ss[r * 81 + i] - mx);
      Ss[r * 81 + i] = e;
      sum += e;
    }
    float inv = 1.f / sum;
    for (int i = 0; i < 80; ++i) Ps[r * 96 + i] = (_Float16)(Ss[r * 81 + i] * inv);
    for (int i = 80; i < 96; ++i) Ps[r * 96 + i] = (_Float16)0.f;
  }
  __syncthreads();

  // O = P V : A from LDS probs, B from LDS V^T, K=96 (zero-padded)
  v8f c0 = {}, c1 = {};
  for (int k0 = 0; k0 < 96; k0 += 32) {
    v16h ap = load_a_frag(&Ps[wave * 16 * 96 + k0], 96, l16, lh);
    v16h b0 = load_b_frag(Vs + k0, 96, l16, lh);
    v16h b1 = load_b_frag(Vs + 16 * 96 + k0, 96, l16, lh);
    c0 = wmma16(ap, b0, c0);
    c1 = wmma16(ap, b1, c1);
  }
  for (int j = 0; j < 8; ++j) {
    int m = wave * 16 + j + lh * 8;
    size_t base = (size_t)(w * 80 + m) * 256 + head * 32;
    o[base + l16]      = (_Float16)c0[j];
    o[base + 16 + l16] = (_Float16)c1[j];
  }
}

// --------------------------- proj GEMM + residual --------------------------
// o[T,256] @ proj_w + b; fused grid-reverse + x + ls1*attn -> x1 (fp32, pixel order)
__global__ void __launch_bounds__(256) k_gemm_proj(const _Float16* __restrict__ A,
                                                   const _Float16* __restrict__ Bt,
                                                   const float* __restrict__ bias,
                                                   const float* __restrict__ x,
                                                   const float* __restrict__ ls1,
                                                   float* __restrict__ x1) {
  int mblk = blockIdx.x * 128, nblk = blockIdx.y * 128;
  v8f c[4][2] = {};
  gemm_core(A + (size_t)mblk * 256, Bt + (size_t)nblk * 256, 256, c);
  int lane = threadIdx.x & 31, l16 = lane & 15, lh = lane >> 4;
  int wid = threadIdx.x >> 5, wm = wid & 1, wn = wid >> 1;
  for (int mf = 0; mf < 4; ++mf)
    for (int nf = 0; nf < 2; ++nf) {
      int n = nblk + wn * 32 + nf * 16 + l16;
      float bv = bias[n], lv = ls1[n];
      for (int j = 0; j < 8; ++j) {
        int t = mblk + wm * 64 + mf * 16 + j + lh * 8;   // window-token index
        float val = c[mf][nf][j] + bv;
        int w = t / 80, nn = t - w * 80;
        int bb = w / 64, rw = w - bb * 64;
        int h2 = rw >> 3, w2 = rw & 7;
        int gh = nn / 10, gw = nn - gh * 10;
        int pix = (bb * 64 + gh * 8 + h2) * 80 + gw * 8 + w2;
        size_t off = (size_t)pix * 256 + n;
        x1[off] = x[off] + lv * val;
      }
    }
}

// --------------------------- fc1 GEMM + GELU -------------------------------
__global__ void __launch_bounds__(256) k_gemm_fc1(const _Float16* __restrict__ A,
                                                  const _Float16* __restrict__ Bt,
                                                  const float* __restrict__ bias,
                                                  _Float16* __restrict__ O) {
  int mblk = blockIdx.x * 128, nblk = blockIdx.y * 128;
  v8f c[4][2] = {};
  gemm_core(A + (size_t)mblk * 256, Bt + (size_t)nblk * 256, 256, c);
  int lane = threadIdx.x & 31, l16 = lane & 15, lh = lane >> 4;
  int wid = threadIdx.x >> 5, wm = wid & 1, wn = wid >> 1;
  for (int mf = 0; mf < 4; ++mf)
    for (int nf = 0; nf < 2; ++nf) {
      int n = nblk + wn * 32 + nf * 16 + l16;
      float bv = bias[n];
      for (int j = 0; j < 8; ++j) {
        int m = mblk + wm * 64 + mf * 16 + j + lh * 8;
        float v = c[mf][nf][j] + bv;
        float gl = 0.5f * v * (1.f + erff(v * 0.70710678118654752f)); // exact GELU
        O[(size_t)m * 1024 + n] = (_Float16)gl;
      }
    }
}

// --------------------------- fc2 GEMM + residual ---------------------------
__global__ void __launch_bounds__(256) k_gemm_fc2(const _Float16* __restrict__ A,
                                                  const _Float16* __restrict__ Bt,
                                                  const float* __restrict__ bias,
                                                  const float* __restrict__ x1,
                                                  const float* __restrict__ ls2,
                                                  float* __restrict__ out) {
  int mblk = blockIdx.x * 128, nblk = blockIdx.y * 128;
  v8f c[4][2] = {};
  gemm_core(A + (size_t)mblk * 1024, Bt + (size_t)nblk * 1024, 1024, c);
  int lane = threadIdx.x & 31, l16 = lane & 15, lh = lane >> 4;
  int wid = threadIdx.x >> 5, wm = wid & 1, wn = wid >> 1;
  for (int mf = 0; mf < 4; ++mf)
    for (int nf = 0; nf < 2; ++nf) {
      int n = nblk + wn * 32 + nf * 16 + l16;
      float bv = bias[n], lv = ls2[n];
      for (int j = 0; j < 8; ++j) {
        int m = mblk + wm * 64 + mf * 16 + j + lh * 8;
        size_t off = (size_t)m * 256 + n;
        out[off] = x1[off] + lv * (c[mf][nf][j] + bv);
      }
    }
}

// ---------------------------------------------------------------------------
extern "C" void kernel_launch(void* const* d_in, const int* in_sizes, int n_in,
                              void* d_out, int out_size, void* d_ws, size_t ws_size,
                              hipStream_t stream) {
  (void)in_sizes; (void)n_in; (void)out_size; (void)ws_size;
  const float* x      = (const float*)d_in[0];
  const float* n1g    = (const float*)d_in[1];
  const float* n1b    = (const float*)d_in[2];
  const float* qkv_w  = (const float*)d_in[3];
  const float* qkv_b  = (const float*)d_in[4];
  const float* proj_w = (const float*)d_in[5];
  const float* proj_b = (const float*)d_in[6];
  const float* ls1    = (const float*)d_in[7];
  const float* n2g    = (const float*)d_in[8];
  const float* n2b    = (const float*)d_in[9];
  const float* fc1_w  = (const float*)d_in[10];
  const float* fc1_b  = (const float*)d_in[11];
  const float* fc2_w  = (const float*)d_in[12];
  const float* fc2_b  = (const float*)d_in[13];
  const float* ls2    = (const float*)d_in[14];
  float* out = (float*)d_out;

  // Workspace layout with lifetime-based reuse (~195 MB total):
  //  R0: h16 (LN1 output, window order)        -> reused as attn output o_buf
  //  R1: q | k (f16)                           -> reused as x1 (fp32, same bytes)
  //  R2: v^T (f16, padded K=96)                -> reused as LN2 output h2
  //  R3: fc1 activation chunk (8192 rows)
  //  R4: transposed f16 weights
  char* ws = (char*)d_ws;
  const size_t T = 81920;
  size_t off = 0;
  _Float16* h16  = (_Float16*)(ws + off); _Float16* obuf = h16; off += T * 256 * 2;
  _Float16* qb   = (_Float16*)(ws + off); float* x1 = (float*)(ws + off); off += T * 256 * 2;
  _Float16* kbuf = (_Float16*)(ws + off); off += T * 256 * 2;
  _Float16* vtb  = (_Float16*)(ws + off); _Float16* h2 = vtb; off += 1024ull * 8 * 32 * 96 * 2;
  _Float16* abuf = (_Float16*)(ws + off); off += 8192ull * 1024 * 2;
  _Float16* qkv_wt  = (_Float16*)(ws + off); off += 768ull * 256 * 2;
  _Float16* proj_wt = (_Float16*)(ws + off); off += 256ull * 256 * 2;
  _Float16* fc1_wt  = (_Float16*)(ws + off); off += 1024ull * 256 * 2;
  _Float16* fc2_wt  = (_Float16*)(ws + off); off += 256ull * 1024 * 2;

  // Weight prep: fp32 [K][N] -> f16 [N][K]
  k_transpose_cast<<<(256 * 768 + 255) / 256, 256, 0, stream>>>(qkv_w, qkv_wt, 256, 768);
  k_transpose_cast<<<(256 * 256 + 255) / 256, 256, 0, stream>>>(proj_w, proj_wt, 256, 256);
  k_transpose_cast<<<(256 * 1024 + 255) / 256, 256, 0, stream>>>(fc1_w, fc1_wt, 256, 1024);
  k_transpose_cast<<<(1024 * 256 + 255) / 256, 256, 0, stream>>>(fc2_w, fc2_wt, 1024, 256);

  // Attention branch
  k_layernorm<<<10240, 256, 0, stream>>>(x, n1g, n1b, h16, 1);
  k_gemm_qkv<<<dim3(640, 6), 256, 0, stream>>>(h16, qkv_wt, qkv_b, qb, kbuf, vtb);
  k_attention<<<8192, 160, 0, stream>>>(qb, kbuf, vtb, obuf);
  k_gemm_proj<<<dim3(640, 2), 256, 0, stream>>>(obuf, proj_wt, proj_b, x, ls1, x1);

  // MLP branch (chunked over rows so the GELU buffer is small)
  k_layernorm<<<10240, 256, 0, stream>>>(x1, n2g, n2b, h2, 0);
  for (int ch = 0; ch < 10; ++ch) {
    const _Float16* Ain = h2 + (size_t)ch * 8192 * 256;
    const float* x1c = x1 + (size_t)ch * 8192 * 256;
    float* oc = out + (size_t)ch * 8192 * 256;
    k_gemm_fc1<<<dim3(64, 8), 256, 0, stream>>>(Ain, fc1_wt, fc1_b, abuf);
    k_gemm_fc2<<<dim3(64, 2), 256, 0, stream>>>(abuf, fc2_wt, fc2_b, x1c, ls2, oc);
  }
}
